// SelfAttention_64304250356186
// MI455X (gfx1250) — matrix operand order repaired and checked by reading, not compiled, not run on hardware
//
#include <hip/hip_runtime.h>

typedef __attribute__((ext_vector_type(16))) _Float16 v16h;
typedef __attribute__((ext_vector_type(8)))  _Float16 v8h;
typedef __attribute__((ext_vector_type(8)))  float    v8f;

#define B_ 2
#define S_ 2048
#define E_ 2048
#define H_ 32
#define D_ 64

static __device__ __forceinline__ v16h cat16(v8h a, v8h b) {
  v16h r;
#pragma unroll
  for (int i = 0; i < 8; ++i) { r[i] = a[i]; r[i + 8] = b[i]; }
  return r;
}
static __device__ __forceinline__ v8h ld8(const _Float16* p) { return *(const v8h*)p; }

static __device__ __forceinline__ v8f wmma_f16(v16h a, v16h b, v8f c) {
  return __builtin_amdgcn_wmma_f32_16x16x32_f16(false, a, false, b, (short)0, c, false, false);
}

// ---------------------------------------------------------------------------
// fp32 -> fp16 conversion
// ---------------------------------------------------------------------------
__global__ void f32_to_f16(const float* __restrict__ s, _Float16* __restrict__ d, int n) {
  int i = blockIdx.x * blockDim.x + threadIdx.x;
  if (i < n) d[i] = (_Float16)s[i];
}

// ---------------------------------------------------------------------------
// C[M][N] = A[M][K] (f16 row-major) * W[N][K]^T (f16 row-major).
// 256 threads (8 waves), 128x128 block tile, K stepped by 64 through
// double-buffered LDS with register prefetch. Waves form a 4x2 grid of
// 32x64 wave tiles: 2 A-frags x 4 B-frags, every B-frag feeds 2 WMMAs.
// ---------------------------------------------------------------------------
template <typename OUT_T>
__global__ __launch_bounds__(256) void gemm_xwt(const _Float16* __restrict__ A,
                                                const _Float16* __restrict__ Bw,
                                                OUT_T* __restrict__ C,
                                                int M, int N, int K) {
  __shared__ _Float16 sA[2][128 * 64];
  __shared__ _Float16 sB[2][128 * 64];
  const int t = threadIdx.x;
  const int wave = t >> 5, lane = t & 31, hf = lane >> 4, l15 = lane & 15;
  const int aklo = hf * 8;    // A-fragment K base ({aklo..aklo+7, 16+aklo..})
  const int bklo = hf * 8;    // B-fragment K base (same order as A)
  const int m0 = blockIdx.y * 128;
  const int n0 = blockIdx.x * 128;
  const int wrow = (wave >> 1) * 32;   // wave tile origin inside block
  const int wcol = (wave & 1) * 64;

  const int ldRow = t >> 1;          // 0..127
  const int ldCol = (t & 1) * 32;    // 0 or 32

  v8f acc[2][4] = {};

  // prologue: stage tile 0
  {
    const _Float16* ga = A  + (size_t)(m0 + ldRow) * K + ldCol;
    const _Float16* gb = Bw + (size_t)(n0 + ldRow) * K + ldCol;
#pragma unroll
    for (int c = 0; c < 4; ++c) {
      *(v8h*)&sA[0][ldRow * 64 + ldCol + c * 8] = ld8(ga + c * 8);
      *(v8h*)&sB[0][ldRow * 64 + ldCol + c * 8] = ld8(gb + c * 8);
    }
  }
  __syncthreads();
  int cur = 0;

  for (int k0 = 0; k0 < K; k0 += 64) {
    v8h pa[4], pb[4];
    const bool more = (k0 + 64) < K;
    if (more) {  // issue next tile's global loads early (latency hidden by WMMA)
      const _Float16* ga = A  + (size_t)(m0 + ldRow) * K + k0 + 64 + ldCol;
      const _Float16* gb = Bw + (size_t)(n0 + ldRow) * K + k0 + 64 + ldCol;
#pragma unroll
      for (int c = 0; c < 4; ++c) { pa[c] = ld8(ga + c * 8); pb[c] = ld8(gb + c * 8); }
    }

    const _Float16* cA = sA[cur];
    const _Float16* cB = sB[cur];
#pragma unroll
    for (int ks = 0; ks < 2; ++ks) {
      const int ar0 = wrow + l15;
      const int ar1 = wrow + 16 + l15;
      v16h av0 = cat16(ld8(&cA[ar0 * 64 + ks * 32 + aklo]),
                       ld8(&cA[ar0 * 64 + ks * 32 + 16 + aklo]));
      v16h av1 = cat16(ld8(&cA[ar1 * 64 + ks * 32 + aklo]),
                       ld8(&cA[ar1 * 64 + ks * 32 + 16 + aklo]));
#pragma unroll
      for (int j = 0; j < 4; ++j) {
        const int br = wcol + j * 16 + l15;
        v16h bv = cat16(ld8(&cB[br * 64 + ks * 32 + bklo]),
                        ld8(&cB[br * 64 + ks * 32 + 16 + bklo]));
        acc[0][j] = wmma_f16(av0, bv, acc[0][j]);
        acc[1][j] = wmma_f16(av1, bv, acc[1][j]);
      }
    }

    if (more) {
      const int nxt = cur ^ 1;
#pragma unroll
      for (int c = 0; c < 4; ++c) {
        *(v8h*)&sA[nxt][ldRow * 64 + ldCol + c * 8] = pa[c];
        *(v8h*)&sB[nxt][ldRow * 64 + ldCol + c * 8] = pb[c];
      }
      __syncthreads();
      cur = nxt;
    }
  }

#pragma unroll
  for (int ai = 0; ai < 2; ++ai) {
    const int crow = m0 + wrow + ai * 16 + hf * 8;
#pragma unroll
    for (int j = 0; j < 4; ++j) {
      const int col = n0 + wcol + j * 16 + l15;
#pragma unroll
      for (int i = 0; i < 8; ++i)
        C[(size_t)(crow + i) * N + col] = (OUT_T)acc[ai][j][i];
    }
  }
}

// ---------------------------------------------------------------------------
// RoPE on Q and K (in place, f16 [b][s][h*64+d]); sm_scale folded into Q.
// ---------------------------------------------------------------------------
__global__ void rope_qk(_Float16* __restrict__ q, _Float16* __restrict__ k) {
  int i = blockIdx.x * blockDim.x + threadIdx.x;  // B*S*H*32 threads
  int d = i & 31;
  int h = (i >> 5) & 31;
  int s = (i >> 10) & (S_ - 1);
  int b = i >> 21;
  float freq = (float)s * __expf(-(float)d * (9.210340371976184f / 32.0f));
  float sn, cs;
  __sincosf(freq, &sn, &cs);
  size_t base = ((size_t)(b * S_ + s)) * E_ + h * 64 + d;
  const float qscale = 0.125f;  // 1/sqrt(D)
  float q1 = (float)q[base], q2 = (float)q[base + 32];
  q[base]      = (_Float16)((q1 * cs - q2 * sn) * qscale);
  q[base + 32] = (_Float16)((q2 * cs + q1 * sn) * qscale);
  float k1 = (float)k[base], k2 = (float)k[base + 32];
  k[base]      = (_Float16)(k1 * cs - k2 * sn);
  k[base + 32] = (_Float16)(k2 * cs + k1 * sn);
}

// ---------------------------------------------------------------------------
// Flash attention: block = 4 waves = 64 query rows of one (b,h).
// 64 keys per iteration through double-buffered LDS; online softmax.
// ---------------------------------------------------------------------------
__global__ __launch_bounds__(128) void flash_attn(const _Float16* __restrict__ qh,
                                                  const _Float16* __restrict__ kh,
                                                  const _Float16* __restrict__ vh,
                                                  const float* __restrict__ mask,
                                                  _Float16* __restrict__ oh) {
  __shared__ _Float16 sK[2][64 * 64];    // [key][d] row-major
  __shared__ _Float16 sVt[2][64 * 64];   // [d][key] (transposed at staging)
  __shared__ _Float16 sP[4 * 16 * 64];   // per-wave P transpose scratch

  const int t = threadIdx.x;
  const int wave = t >> 5, lane = t & 31, hf = lane >> 4, l15 = lane & 15;
  const int aklo = hf * 8, bklo = hf * 8;
  const int b = blockIdx.z, h = blockIdx.y;
  const int q0 = blockIdx.x * 64 + wave * 16;

  // staging coordinates: 64 keys x 64 d, 128 threads, 4x16B per thread
  const int skey = t >> 1;          // 0..63
  const int sdb  = (t & 1) * 32;    // 0 or 32

  // Q fragments (A layout), d 0..31 and 32..63.
  const _Float16* qp = qh + ((size_t)(b * S_) + q0 + l15) * E_ + h * 64;
  v16h qa0 = cat16(ld8(qp + aklo),      ld8(qp + 16 + aklo));
  v16h qa1 = cat16(ld8(qp + 32 + aklo), ld8(qp + 48 + aklo));

  float m[8], l[8];
  v8f o[4] = {};
#pragma unroll
  for (int i = 0; i < 8; ++i) { m[i] = -3.0e38f; l[i] = 0.f; }

  // prologue: stage keys 0..63
  {
    size_t g = ((size_t)(b * S_) + skey) * E_ + h * 64 + sdb;
#pragma unroll
    for (int c = 0; c < 4; ++c) {
      v8h kv = ld8(kh + g + c * 8);
      v8h vv = ld8(vh + g + c * 8);
      *(v8h*)&sK[0][skey * 64 + sdb + c * 8] = kv;
#pragma unroll
      for (int i2 = 0; i2 < 8; ++i2) sVt[0][(sdb + c * 8 + i2) * 64 + skey] = vv[i2];
    }
  }
  __syncthreads();
  int cur = 0;

  for (int kk = 0; kk < S_; kk += 64) {
    v8h pk[4], pv[4];
    const bool more = (kk + 64) < S_;
    if (more) {  // prefetch next 64-key tile
      size_t g = ((size_t)(b * S_) + kk + 64 + skey) * E_ + h * 64 + sdb;
#pragma unroll
      for (int c = 0; c < 4; ++c) { pk[c] = ld8(kh + g + c * 8); pv[c] = ld8(vh + g + c * 8); }
    }

    const _Float16* K  = sK[cur];
    const _Float16* Vt = sVt[cur];

    // scores: four 16x16 fragments over 64 keys
    v8f sc4[4];
#pragma unroll
    for (int j = 0; j < 4; ++j) {
      const int kr = j * 16 + l15;
      v16h b0 = cat16(ld8(&K[kr * 64 + bklo]),      ld8(&K[kr * 64 + 16 + bklo]));
      v16h b1 = cat16(ld8(&K[kr * 64 + 32 + bklo]), ld8(&K[kr * 64 + 48 + bklo]));
      v8f s = {};
      s = wmma_f16(qa0, b0, s);
      s = wmma_f16(qa1, b1, s);
      sc4[j] = s;
    }
    float mv[4];
#pragma unroll
    for (int j = 0; j < 4; ++j) mv[j] = mask[b * S_ + kk + j * 16 + l15];

    // online softmax; one butterfly per row covers all 64 columns
    _Float16* pw = &sP[wave * 1024];
#pragma unroll
    for (int i = 0; i < 8; ++i) {
      float x[4];
      float rm = -3.0e38f;
#pragma unroll
      for (int j = 0; j < 4; ++j) { x[j] = sc4[j][i] + mv[j]; rm = fmaxf(rm, x[j]); }
#pragma unroll
      for (int msk = 1; msk < 16; msk <<= 1) rm = fmaxf(rm, __shfl_xor(rm, msk));
      float mn = fmaxf(m[i], rm);
      float scl = __expf(m[i] - mn);
      m[i] = mn;
      float rs = 0.f;
#pragma unroll
      for (int j = 0; j < 4; ++j) { x[j] = __expf(x[j] - mn); rs += x[j]; }
#pragma unroll
      for (int msk = 1; msk < 16; msk <<= 1) rs += __shfl_xor(rs, msk);
      l[i] = l[i] * scl + rs;
#pragma unroll
      for (int j2 = 0; j2 < 4; ++j2) o[j2][i] *= scl;
#pragma unroll
      for (int j = 0; j < 4; ++j)
        pw[(i + 8 * hf) * 64 + j * 16 + l15] = (_Float16)x[j];
    }

    asm volatile("s_wait_dscnt 0" ::: "memory");
    // P in A-layout: two fragments over the 64-key K-dimension
    v16h pa0 = cat16(ld8(&pw[l15 * 64 + aklo]),      ld8(&pw[l15 * 64 + 16 + aklo]));
    v16h pa1 = cat16(ld8(&pw[l15 * 64 + 32 + aklo]), ld8(&pw[l15 * 64 + 48 + aklo]));
#pragma unroll
    for (int j2 = 0; j2 < 4; ++j2) {
      const int d = j2 * 16 + l15;
      v16h bv0 = cat16(ld8(&Vt[d * 64 + bklo]),      ld8(&Vt[d * 64 + 16 + bklo]));
      v16h bv1 = cat16(ld8(&Vt[d * 64 + 32 + bklo]), ld8(&Vt[d * 64 + 48 + bklo]));
      o[j2] = wmma_f16(pa0, bv0, o[j2]);
      o[j2] = wmma_f16(pa1, bv1, o[j2]);
    }

    if (more) {
      const int nxt = cur ^ 1;
#pragma unroll
      for (int c = 0; c < 4; ++c) {
        *(v8h*)&sK[nxt][skey * 64 + sdb + c * 8] = pk[c];
#pragma unroll
        for (int i2 = 0; i2 < 8; ++i2) sVt[nxt][(sdb + c * 8 + i2) * 64 + skey] = pv[c][i2];
      }
      __syncthreads();
      cur = nxt;
    }
  }

  float inv[8];
#pragma unroll
  for (int i = 0; i < 8; ++i) inv[i] = 1.f / l[i];
  const int orow = q0 + hf * 8;
#pragma unroll
  for (int j = 0; j < 4; ++j)
#pragma unroll
    for (int i = 0; i < 8; ++i)
      oh[((size_t)(b * S_) + orow + i) * E_ + h * 64 + j * 16 + l15] =
          (_Float16)(o[j][i] * inv[i]);
}

// ---------------------------------------------------------------------------
extern "C" void kernel_launch(void* const* d_in, const int* in_sizes, int n_in,
                              void* d_out, int out_size, void* d_ws, size_t ws_size,
                              hipStream_t stream) {
  (void)in_sizes; (void)n_in; (void)out_size; (void)ws_size;
  const float* x  = (const float*)d_in[0];
  const float* mk = (const float*)d_in[1];
  const float* Wq = (const float*)d_in[2];
  const float* Wk = (const float*)d_in[3];
  const float* Wv = (const float*)d_in[4];
  const float* Wo = (const float*)d_in[5];
  float* out = (float*)d_out;

  const size_t MROWS = (size_t)B_ * S_;     // 4096
  const size_t EE    = (size_t)E_ * E_;     // 4.2M
  const size_t XSZ   = MROWS * E_;          // 8.4M

  _Float16* xh  = (_Float16*)d_ws;
  _Float16* wqh = xh + XSZ;
  _Float16* wkh = wqh + EE;
  _Float16* wvh = wkh + EE;
  _Float16* woh = wvh + EE;
  _Float16* qh  = woh + EE;
  _Float16* kh  = qh + XSZ;
  _Float16* vh  = kh + XSZ;
  _Float16* ah  = vh + XSZ;

  f32_to_f16<<<(int)(XSZ / 256), 256, 0, stream>>>(x, xh, (int)XSZ);
  f32_to_f16<<<(int)(EE / 256), 256, 0, stream>>>(Wq, wqh, (int)EE);
  f32_to_f16<<<(int)(EE / 256), 256, 0, stream>>>(Wk, wkh, (int)EE);
  f32_to_f16<<<(int)(EE / 256), 256, 0, stream>>>(Wv, wvh, (int)EE);
  f32_to_f16<<<(int)(EE / 256), 256, 0, stream>>>(Wo, woh, (int)EE);

  dim3 gg(E_ / 128, (unsigned)(MROWS / 128));
  gemm_xwt<_Float16><<<gg, 256, 0, stream>>>(xh, wqh, qh, (int)MROWS, E_, E_);
  gemm_xwt<_Float16><<<gg, 256, 0, stream>>>(xh, wkh, kh, (int)MROWS, E_, E_);
  gemm_xwt<_Float16><<<gg, 256, 0, stream>>>(xh, wvh, vh, (int)MROWS, E_, E_);

  rope_qk<<<(B_ * S_ * H_ * 32) / 256, 256, 0, stream>>>(qh, kh);

  flash_attn<<<dim3(S_ / 64, H_, B_), 128, 0, stream>>>(qh, kh, vh, mk, ah);

  gemm_xwt<float><<<gg, 256, 0, stream>>>(ah, woh, out, (int)MROWS, E_, E_);
}
